// VisionAttention_78726750536004
// MI455X (gfx1250) — matrix operand order repaired and checked by reading, not compiled, not run on hardware
//
#include <hip/hip_runtime.h>
#include <hip/hip_bf16.h>
#include <stdint.h>

// ---------------------------------------------------------------------------
// Vision attention (B=1, L=4096, D=1152, N=16 heads, H=72, S=64 grid) for
// MI455X / gfx1250.  All GEMM-like math via v_wmma_f32_16x16x32_bf16 (wave32),
// f32 accumulate.  256-thread (8-wave) workgroups with double-buffered LDS
// staging; per-wave tiles sized so register pressure stays under 256 VGPRs
// (round-2 disassembly showed accumulator spills to scratch with 64x64/wave).
//   - GEMM: 128x128 block per WG, 32x64 per wave (64 acc VGPRs)
//   - attention: 128 queries + one K/V stage per WG, 16 queries per wave
// ---------------------------------------------------------------------------

typedef __bf16 bf16_t;
typedef __attribute__((ext_vector_type(16))) __bf16 v16bf;
typedef __attribute__((ext_vector_type(8)))  __bf16 v8bf;
typedef __attribute__((ext_vector_type(8)))  float  v8f;

namespace cfg {
constexpr int L   = 4096;
constexpr int D   = 1152;
constexpr int NH  = 16;
constexpr int H   = 72;
constexpr int HP  = 96;       // head dim padded to 3x32 WMMA K-chunks
constexpr int D3  = 3 * D;    // fused QKV output width (3456)
}

// ---- workspace layout (bytes, 256-aligned) --------------------------------
constexpr size_t OFF_XB     = 0;                                   // bf16 [L][D]
constexpr size_t SZ_XB      = (size_t)cfg::L * cfg::D * 2;
constexpr size_t OFF_WALLT  = OFF_XB + SZ_XB;                      // bf16 [3456][1152] (Wq^T ++ Wkv^T)
constexpr size_t SZ_WALLT   = (size_t)cfg::D3 * cfg::D * 2;
constexpr size_t OFF_WOT    = OFF_WALLT + SZ_WALLT;                // bf16 [1152][1152] (Wo^T)
constexpr size_t SZ_WOT     = (size_t)cfg::D * cfg::D * 2;
constexpr size_t OFF_BIAS   = OFF_WOT + SZ_WOT;                    // f32 [3456]
constexpr size_t SZ_BIAS    = (size_t)cfg::D3 * 4 + 256;
constexpr size_t OFF_QKVRAW = OFF_BIAS + SZ_BIAS;                  // f32 [L][3456]
constexpr size_t SZ_QKVRAW  = (size_t)cfg::L * cfg::D3 * 4;
constexpr size_t OFF_QP     = OFF_QKVRAW + SZ_QKVRAW;              // bf16 [N][L][HP]
constexpr size_t SZ_HEADS   = (size_t)cfg::NH * cfg::L * cfg::HP * 2;
constexpr size_t OFF_KP     = OFF_QP + SZ_HEADS;                   // bf16 [N][L][HP]
constexpr size_t OFF_VT     = OFF_KP + SZ_HEADS;                   // bf16 [N][HP][L] (V transposed)
constexpr size_t OFF_OIN    = OFF_VT + SZ_HEADS;                   // bf16 [L][D]

// ---------------------------------------------------------------------------
// Fragment loaders, per CDNA5 ISA §7.12.2 layouts.
//   A (16x32, bf16): lane m (0..15) holds K=0..7 & 16..23; lane m+16 holds
//                    K=8..15 & 24..31 of row m.
//   B (32x16, bf16): lane n (0..15) holds column n, K=0..7 & 16..23;
//                    lane n+16 holds K=8..15 & 24..31.  The B operand is always a
//                    transposed (N-major) buffer so loads are contiguous.
// Works for both global and LDS source pointers.
// ---------------------------------------------------------------------------
__device__ __forceinline__ v16bf load_a_frag(const bf16_t* row_kc, int hi) {
  const v8bf* p = (const v8bf*)row_kc;           // 8-elem (16B) chunks of K
  v8bf lo  = p[hi];                              // K =  hi*8 .. +8
  v8bf hi8 = p[2 + hi];                          // K = 16 + hi*8 .. +8
  return __builtin_shufflevector(lo, hi8, 0, 1, 2, 3, 4, 5, 6, 7,
                                          8, 9, 10, 11, 12, 13, 14, 15);
}

__device__ __forceinline__ v16bf load_b_frag(const bf16_t* colrow_kc, int hi) {
  const v8bf* p = (const v8bf*)colrow_kc;        // K = hi*8..+8, 16+hi*8..+8
  v8bf lo  = p[hi];
  v8bf hi8 = p[2 + hi];
  return __builtin_shufflevector(lo, hi8, 0, 1, 2, 3, 4, 5, 6, 7,
                                          8, 9, 10, 11, 12, 13, 14, 15);
}

// ---------------------------------------------------------------------------
// Prep kernels
// ---------------------------------------------------------------------------
__global__ void cast_bf16_kernel(const float* __restrict__ src,
                                 bf16_t* __restrict__ dst, int n) {
  int i = blockIdx.x * blockDim.x + threadIdx.x;
  if (i < n) dst[i] = (bf16_t)src[i];
}

// dst[o][i] = src[i][o]; dst rows have length rows_in.
__global__ void transpose_cast_kernel(const float* __restrict__ src,
                                      bf16_t* __restrict__ dst,
                                      int rows_in, int cols_out) {
  int idx = blockIdx.x * blockDim.x + threadIdx.x;
  if (idx >= rows_in * cols_out) return;
  int o = idx / rows_in;
  int i = idx - o * rows_in;
  dst[(size_t)o * rows_in + i] = (bf16_t)src[(size_t)i * cols_out + o];
}

__global__ void bias_concat_kernel(const float* __restrict__ qb,
                                   const float* __restrict__ kvb,
                                   float* __restrict__ dst) {
  int i = blockIdx.x * blockDim.x + threadIdx.x;
  if (i < cfg::D3) dst[i] = (i < cfg::D) ? qb[i] : kvb[i - cfg::D];
}

// ---------------------------------------------------------------------------
// WMMA GEMM:  C[M][Ntot] = A[M][K](bf16) x BT[Ntot][K](bf16)^T + bias
// 8-wave (256-thread) workgroup computes a 128x128 block; per 32-K-chunk the
// WG stages A(128x32) and B(128x32) into double-buffered LDS; each wave
// computes a 32x64 sub-tile (2x4 WMMA accumulators = 64 VGPRs, no spills).
// ---------------------------------------------------------------------------
__global__ void gemm_bias_kernel(const bf16_t* __restrict__ A,
                                 const bf16_t* __restrict__ BT,
                                 const float* __restrict__ bias,
                                 float* __restrict__ C,
                                 int Ntot, int K) {
  const int tid  = threadIdx.x;                  // 256 threads = 8 waves
  const int w    = tid >> 5;
  const int lane = tid & 31;
  const int m16  = lane & 15;
  const int hi   = lane >> 4;
  const int mbase = blockIdx.x * 128;
  const int nbase = blockIdx.y * 128;
  const int wm = (w >> 1) * 32;                  // wave sub-tile: rows (4x32)
  const int wn = (w & 1) * 64;                   //                cols (2x64)

  __shared__ __align__(16) bf16_t a_lds[2][128][32];
  __shared__ __align__(16) bf16_t b_lds[2][128][32];

  v8f acc[2][4];
#pragma unroll
  for (int i = 0; i < 2; ++i)
#pragma unroll
    for (int j = 0; j < 4; ++j) acc[i][j] = (v8f){0, 0, 0, 0, 0, 0, 0, 0};

  // stage one 32-wide K-chunk of A and B into LDS buffer `buf`
  auto stage = [&](int buf, int kc) {
    for (int c = tid; c < 512; c += 256) {       // 128 rows x 4 chunks of 16B
      int r   = c >> 2;
      int off = (c & 3) * 8;
      *(uint4*)&a_lds[buf][r][off] =
          *(const uint4*)(A + (size_t)(mbase + r) * K + kc + off);
      *(uint4*)&b_lds[buf][r][off] =
          *(const uint4*)(BT + (size_t)(nbase + r) * K + kc + off);
    }
  };

  stage(0, 0);
  int buf = 0;
  for (int kc = 0; kc < K; kc += 32, buf ^= 1) {
    __syncthreads();                             // staged chunk visible
    if (kc + 32 < K) stage(buf ^ 1, kc + 32);    // overlap next stage

    v16bf af[2], bfv[4];
#pragma unroll
    for (int g = 0; g < 2; ++g)
      af[g] = load_a_frag(&a_lds[buf][wm + g * 16 + m16][0], hi);
#pragma unroll
    for (int g = 0; g < 4; ++g)
      bfv[g] = load_b_frag(&b_lds[buf][wn + g * 16 + m16][0], hi);
#pragma unroll
    for (int i = 0; i < 2; ++i)
#pragma unroll
      for (int j = 0; j < 4; ++j)
        acc[i][j] = __builtin_amdgcn_wmma_f32_16x16x32_bf16(
            false, af[i], false, bfv[j], (short)0, acc[i][j], false, false);
  }

#pragma unroll
  for (int tn = 0; tn < 4; ++tn) {
    const int col = nbase + wn + tn * 16 + m16;
    const float b = bias[col];
#pragma unroll
    for (int tm = 0; tm < 2; ++tm) {
#pragma unroll
      for (int j = 0; j < 8; ++j) {
        int r = mbase + wm + tm * 16 + j + hi * 8;   // C layout: row = j+hi*8
        C[(size_t)r * Ntot + col] = acc[tm][tn][j] + b;
      }
    }
  }
}

// ---------------------------------------------------------------------------
// RMSNorm + 2-axis RoPE + pad/layout for attention.
// One block (256 threads) per token l; row of 3456 f32 staged in LDS.
// ---------------------------------------------------------------------------
__global__ void normrope_kernel(const float* __restrict__ qkvraw,
                                const float* __restrict__ qscale,
                                const float* __restrict__ kscale,
                                bf16_t* __restrict__ qp,
                                bf16_t* __restrict__ kp,
                                bf16_t* __restrict__ vt) {
  using namespace cfg;
  const int l   = blockIdx.x;
  const int tid = threadIdx.x;

  __shared__ float row[3 * D];
  __shared__ float rrms[3][NH];

  const float* src = qkvraw + (size_t)l * D3;
  for (int i = tid; i < D3; i += 256) row[i] = src[i];
  __syncthreads();

  const int wave = tid >> 5, lane = tid & 31;
  for (int item = wave; item < 48; item += 8) {  // 3 tensors x 16 heads
    const int tsel = item / NH;
    const int n    = item - tsel * NH;
    const float* t = &row[tsel * D + n * H];
    float ss = 0.f;
    for (int i = lane; i < H; i += 32) { float v = t[i]; ss += v * v; }
    for (int off = 1; off < 32; off <<= 1) ss += __shfl_xor(ss, off, 32);
    if (lane == 0) rrms[tsel][n] = rsqrtf(ss / (float)H + 1e-6f);
  }
  __syncthreads();

  const float rowf = (float)(l >> 6);            // rows = l / 64
  const float colf = (float)(l & 63);            // cols = l % 64

  for (int idx = tid; idx < NH * HP; idx += 256) {
    const int n = idx / HP;
    const int h = idx - n * HP;

    float vv = 0.f;
    if (h < H) vv = row[2 * D + n * H + h] * rrms[2][n];
    vt[((size_t)n * HP + h) * L + l] = (bf16_t)vv;

    float qv = 0.f, kv = 0.f;
    if (h < H) {
      const int   a   = (h < 36) ? 0 : 36;       // axis offset within head
      const float pos = (h < 36) ? rowf : colf;
      const int   hh  = h - a;
      const int   i   = (hh < 18) ? hh : hh - 18;
      const float freq = __powf(100.0f, -(float)i / 9.0f);
      const float ang  = pos * freq;
      const float cs = __cosf(ang), sn = __sinf(ang);

      const float rq = rrms[0][n], rk = rrms[1][n];
      const float q1 = row[n * H + a + i]      * rq * (1.f + qscale[a + i]);
      const float q2 = row[n * H + a + 18 + i] * rq * (1.f + qscale[a + 18 + i]);
      const float k1 = row[D + n * H + a + i]      * rk * (1.f + kscale[a + i]);
      const float k2 = row[D + n * H + a + 18 + i] * rk * (1.f + kscale[a + 18 + i]);
      if (hh < 18) { qv = q1 * cs - q2 * sn; kv = k1 * cs - k2 * sn; }
      else         { qv = q1 * sn + q2 * cs; kv = k1 * sn + k2 * cs; }
    }
    qp[((size_t)n * L + l) * HP + h] = (bf16_t)qv;
    kp[((size_t)n * L + l) * HP + h] = (bf16_t)kv;
  }
}

// ---------------------------------------------------------------------------
// Flash attention.  8-wave (256-thread) WG handles 128 queries of one head
// (16 per wave).  Per 32-key step the WG stages K (32xHP) and V^T (HPx32)
// into double-buffered LDS once; global_prefetch pulls the tile two steps
// ahead.  Online softmax stats live per-lane in the WMMA C-layout; P is
// re-swizzled via per-wave LDS into the A-layout.
// ---------------------------------------------------------------------------
__global__ void attn_kernel(const bf16_t* __restrict__ qp,
                            const bf16_t* __restrict__ kp,
                            const bf16_t* __restrict__ vt,
                            bf16_t* __restrict__ oin) {
  using namespace cfg;
  const int tid  = threadIdx.x;                  // 256 threads = 8 waves
  const int w    = tid >> 5;
  const int lane = tid & 31;
  const int m16  = lane & 15;
  const int hi   = lane >> 4;
  const int n    = blockIdx.y;
  const int qtbase = blockIdx.x * 128 + w * 16;  // this wave's 16 queries

  const bf16_t* qhead = qp + (size_t)n * L * HP;
  const bf16_t* khead = kp + (size_t)n * L * HP;
  const bf16_t* vhead = vt + (size_t)n * HP * L;

  __shared__ __align__(16) bf16_t k_lds[2][32][HP];   // [key][h]
  __shared__ __align__(16) bf16_t v_lds[2][HP][32];   // [h][key]
  __shared__ __align__(64) bf16_t plds[8][16][32];    // per-wave P swizzle

  // Q A-fragments (3 K-chunks over HP=96), loaded once.
  v16bf qf[3];
  {
    const bf16_t* qrow = qhead + (size_t)(qtbase + m16) * HP;
#pragma unroll
    for (int c = 0; c < 3; ++c) qf[c] = load_a_frag(qrow + c * 32, hi);
  }

  v8f acc[6];
#pragma unroll
  for (int t = 0; t < 6; ++t) acc[t] = (v8f){0, 0, 0, 0, 0, 0, 0, 0};
  float mrow[8], lrow[8];
#pragma unroll
  for (int j = 0; j < 8; ++j) { mrow[j] = -3.0e38f; lrow[j] = 0.f; }

  // cooperative stage of one 32-key K/V tile into LDS buffer `buf`
  auto stage = [&](int buf, int kb) {
    const bf16_t* ksrc = khead + (size_t)kb * HP;
    for (int c = tid; c < 384; c += 256) {       // 32 rows x 12 chunks of 16B
      int key = c / 12, off = (c - key * 12) * 8;
      *(uint4*)&k_lds[buf][key][off] =
          *(const uint4*)(ksrc + (size_t)key * HP + off);
    }
    const bf16_t* vsrc = vhead + kb;
    for (int c = tid; c < 384; c += 256) {       // 96 rows x 4 chunks of 16B
      int h = c >> 2, off = (c & 3) * 8;
      *(uint4*)&v_lds[buf][h][off] =
          *(const uint4*)(vsrc + (size_t)h * L + off);
    }
  };

  stage(0, 0);
  int buf = 0;
  for (int kb = 0; kb < L; kb += 32, buf ^= 1) {
    // prefetch two tiles ahead (global_prefetch_b8)
    if (kb + 64 < L) {
      __builtin_prefetch(khead + (size_t)(kb + 64) * HP + tid * 12, 0, 0);
      if (tid < HP) __builtin_prefetch(vhead + (size_t)tid * L + kb + 64, 0, 0);
    }
    __syncthreads();                             // staged tile visible
    if (kb + 32 < L) stage(buf ^ 1, kb + 32);    // overlap next stage

    // ---- S = Q K^T for 32 keys (two 16x16 tiles) ----
    v8f s0 = (v8f){0, 0, 0, 0, 0, 0, 0, 0};
    v8f s1 = (v8f){0, 0, 0, 0, 0, 0, 0, 0};
#pragma unroll
    for (int c = 0; c < 3; ++c) {
      v16bf kf0 = load_b_frag(&k_lds[buf][m16][c * 32], hi);
      v16bf kf1 = load_b_frag(&k_lds[buf][16 + m16][c * 32], hi);
      s0 = __builtin_amdgcn_wmma_f32_16x16x32_bf16(
          false, qf[c], false, kf0, (short)0, s0, false, false);
      s1 = __builtin_amdgcn_wmma_f32_16x16x32_bf16(
          false, qf[c], false, kf1, (short)0, s1, false, false);
    }

    // ---- online softmax (row r = j + hi*8, cols striped over 16 lanes) ----
#pragma unroll
    for (int j = 0; j < 8; ++j) {
      float tmax = fmaxf(s0[j], s1[j]);
      for (int off = 1; off < 16; off <<= 1)
        tmax = fmaxf(tmax, __shfl_xor(tmax, off, 32));
      const float mnew = fmaxf(mrow[j], tmax);
      const float corr = __expf(mrow[j] - mnew);
      mrow[j] = mnew;
      const float p0 = __expf(s0[j] - mnew);
      const float p1 = __expf(s1[j] - mnew);
      s0[j] = p0; s1[j] = p1;
      float rs = p0 + p1;
      for (int off = 1; off < 16; off <<= 1) rs += __shfl_xor(rs, off, 32);
      lrow[j] = lrow[j] * corr + rs;
#pragma unroll
      for (int t = 0; t < 6; ++t) acc[t][j] *= corr;
    }

    // ---- re-swizzle P (C-layout -> A-layout) via per-wave LDS ----
#pragma unroll
    for (int j = 0; j < 8; ++j) {
      plds[w][j + hi * 8][m16]      = (bf16_t)s0[j];
      plds[w][j + hi * 8][16 + m16] = (bf16_t)s1[j];
    }
    v16bf pf = load_a_frag(&plds[w][m16][0], hi);    // same-wave DS, in-order

    // ---- O += P V ----
#pragma unroll
    for (int t = 0; t < 6; ++t) {
      v16bf vf = load_b_frag(&v_lds[buf][t * 16 + m16][0], hi);
      acc[t] = __builtin_amdgcn_wmma_f32_16x16x32_bf16(
          false, pf, false, vf, (short)0, acc[t], false, false);
    }
  }

  // ---- normalize and scatter to [L][D] bf16 (drop padded h >= 72) ----
#pragma unroll
  for (int t = 0; t < 6; ++t) {
    const int h = t * 16 + m16;
    if (h < H) {
#pragma unroll
      for (int j = 0; j < 8; ++j) {
        const int q = qtbase + j + hi * 8;
        oin[(size_t)q * D + n * H + h] = (bf16_t)(acc[t][j] / lrow[j]);
      }
    }
  }
}

// ---------------------------------------------------------------------------
// Launch
// ---------------------------------------------------------------------------
extern "C" void kernel_launch(void* const* d_in, const int* in_sizes, int n_in,
                              void* d_out, int out_size, void* d_ws, size_t ws_size,
                              hipStream_t stream) {
  using namespace cfg;
  const float* x       = (const float*)d_in[0];
  const float* q_w     = (const float*)d_in[1];
  const float* q_b     = (const float*)d_in[2];
  const float* kv_w    = (const float*)d_in[3];
  const float* kv_b    = (const float*)d_in[4];
  const float* o_w     = (const float*)d_in[5];
  const float* o_b     = (const float*)d_in[6];
  const float* q_scale = (const float*)d_in[7];
  const float* k_scale = (const float*)d_in[8];

  uint8_t* ws = (uint8_t*)d_ws;
  bf16_t* xb      = (bf16_t*)(ws + OFF_XB);
  bf16_t* wallT   = (bf16_t*)(ws + OFF_WALLT);
  bf16_t* woT     = (bf16_t*)(ws + OFF_WOT);
  float*  biasqkv = (float*) (ws + OFF_BIAS);
  float*  qkvraw  = (float*) (ws + OFF_QKVRAW);
  bf16_t* qp      = (bf16_t*)(ws + OFF_QP);
  bf16_t* kp      = (bf16_t*)(ws + OFF_KP);
  bf16_t* vt      = (bf16_t*)(ws + OFF_VT);
  bf16_t* oin     = (bf16_t*)(ws + OFF_OIN);

  // 1) casts / transposed bf16 weights / fused bias
  {
    int n = L * D;
    cast_bf16_kernel<<<(n + 255) / 256, 256, 0, stream>>>(x, xb, n);
  }
  {
    int n = D * D;
    transpose_cast_kernel<<<(n + 255) / 256, 256, 0, stream>>>(q_w, wallT, D, D);
  }
  {
    int n = D * 2 * D;
    transpose_cast_kernel<<<(n + 255) / 256, 256, 0, stream>>>(
        kv_w, wallT + (size_t)D * D, D, 2 * D);
  }
  {
    int n = D * D;
    transpose_cast_kernel<<<(n + 255) / 256, 256, 0, stream>>>(o_w, woT, D, D);
  }
  bias_concat_kernel<<<(D3 + 255) / 256, 256, 0, stream>>>(q_b, kv_b, biasqkv);

  // 2) fused QKV projection (WMMA, LDS-staged 128x128 blocks)
  gemm_bias_kernel<<<dim3(L / 128, D3 / 128), 256, 0, stream>>>(
      xb, wallT, biasqkv, qkvraw, D3, D);

  // 3) rmsnorm + 2D rope + pad/transpose into attention layouts
  normrope_kernel<<<L, 256, 0, stream>>>(qkvraw, q_scale, k_scale, qp, kp, vt);

  // 4) flash attention (WMMA), 128 queries per 8-wave WG per head
  attn_kernel<<<dim3(L / 128, NH), 256, 0, stream>>>(qp, kp, vt, oin);

  // 5) output projection (WMMA) -> f32 d_out
  gemm_bias_kernel<<<dim3(L / 128, D / 128), 256, 0, stream>>>(
      oin, woT, o_b, (float*)d_out, D, D);
}
